// TransformerMeantimeBlock_3384434229566
// MI455X (gfx1250) — matrix operand order repaired and checked by reading, not compiled, not run on hardware
//
#include <hip/hip_runtime.h>
#include <hip/hip_bf16.h>

#define B_   16
#define T_   128
#define D_   256
#define LA   2
#define LR   2
#define H_   4
#define DK_  64
#define DFF_ 1024
#define SCALE_ 0.125f   /* 1/sqrt(64) */
#define EPS_ 1e-5f

typedef __attribute__((ext_vector_type(16))) __bf16 v16bf;
typedef __attribute__((ext_vector_type(8)))  __bf16 v8bf;
typedef __attribute__((ext_vector_type(8)))  float  v8f;

__device__ __forceinline__ float wave_sum(float v) {
#pragma unroll
  for (int o = 16; o > 0; o >>= 1) v += __shfl_xor(v, o, 32);
  return v;
}
__device__ __forceinline__ float wave_max(float v) {
#pragma unroll
  for (int o = 16; o > 0; o >>= 1) v = fmaxf(v, __shfl_xor(v, o, 32));
  return v;
}
__device__ __forceinline__ float gelu_tanh(float x) {
  const float k0 = 0.7978845608028654f;
  const float k1 = 0.044715f;
  return 0.5f * x * (1.0f + tanhf(k0 * (x + k1 * x * x * x)));
}

// A-fragment (16x32 bf16): lane -> M row; per lane two contiguous 16B chunks
// at K = hlf*8 and K = 16 + hlf*8 (ISA 16-bit A layout).
__device__ __forceinline__ v16bf make_afrag(const __bf16* rowp, int hlf) {
  v8bf lo = *(const v8bf*)(rowp + hlf * 8);
  v8bf hi = *(const v8bf*)(rowp + 16 + hlf * 8);
  v16bf a;
#pragma unroll
  for (int t = 0; t < 8; ++t) { a[t] = lo[t]; a[t + 8] = hi[t]; }
  return a;
}
// B-fragment (32x16 bf16): lane -> N column; K = hlf*8+t and 16+hlf*8+t (stride ld).
__device__ __forceinline__ v16bf make_bfrag(const __bf16* p, int ld, int hlf) {
  v16bf b;
#pragma unroll
  for (int t = 0; t < 8; ++t) { b[t] = p[(hlf * 8 + t) * ld]; b[t + 8] = p[(16 + hlf * 8 + t) * ld]; }
  return b;
}

// ---------------------------------------------------------------------------
// Generic GEMM: C[M,N] = act(A[M,K] * B + bias) (+ res).
// Register-staged + LDS double-buffered pipeline: global loads for tile t+1
// are issued (no wait) before the WMMA work of tile t, so HBM latency hides
// under matrix math. One barrier per K-step.
// Block = 128 threads (4 waves), tile 64x64, K-step 32, bf16 WMMA.
// Requires M%64==0, N%64==0, K%32==0 (true for all shapes here).
// ---------------------------------------------------------------------------
#define SA_G 40   /* A LDS row stride (bf16), 80B: conflict-free, 16B aligned */
#define SB_G 72   /* B LDS row stride (bf16), 144B */

template <bool VECB>
__global__ __launch_bounds__(128)
void gemm_bf16_wmma(const float* __restrict__ A, int lda,
                    const float* __restrict__ Bm, long bks, long bns,
                    const float* __restrict__ bias,
                    const float* __restrict__ res,
                    float* __restrict__ C, int ldc,
                    int K, int act) {
  __shared__ __bf16 As[2][64 * SA_G];
  __shared__ __bf16 Bs[2][32 * SB_G];

  const int tid = threadIdx.x;
  const int m0 = blockIdx.y * 64;
  const int n0 = blockIdx.x * 64;
  const int wid = tid >> 5, lane = tid & 31;
  const int hlf = lane >> 4, ln = lane & 15;
  const int wm = (wid >> 1) * 32, wn = (wid & 1) * 32;

  v8f acc[2][2] = {};

  float4 ar[4];          // staged A values (fp32, in flight)
  float4 brv[4];         // staged B values (vector path)
  float  bsc[16];        // staged B values (strided path)

  auto load_tile = [&](int kt) {
#pragma unroll
    for (int it = 0; it < 4; ++it) {
      int idx = tid + it * 128;
      int r = idx >> 3, c4 = idx & 7;
      ar[it] = *(const float4*)(A + (size_t)(m0 + r) * lda + kt + c4 * 4);
    }
    if constexpr (VECB) {
#pragma unroll
      for (int it = 0; it < 4; ++it) {
        int idx = tid + it * 128;
        int r = idx >> 4, c4 = idx & 15;
        brv[it] = *(const float4*)(Bm + (size_t)(kt + r) * bks + n0 + c4 * 4);
      }
    } else {
#pragma unroll
      for (int it = 0; it < 16; ++it) {
        int idx = tid + it * 128;
        int r = idx >> 6, c = idx & 63;
        bsc[it] = Bm[(size_t)(kt + r) * bks + (size_t)(n0 + c) * bns];
      }
    }
  };

  auto store_tile = [&](int buf) {
#pragma unroll
    for (int it = 0; it < 4; ++it) {
      int idx = tid + it * 128;
      int r = idx >> 3, c4 = idx & 7;
      __bf16* d = &As[buf][r * SA_G + c4 * 4];
      d[0] = (__bf16)ar[it].x; d[1] = (__bf16)ar[it].y;
      d[2] = (__bf16)ar[it].z; d[3] = (__bf16)ar[it].w;
    }
    if constexpr (VECB) {
#pragma unroll
      for (int it = 0; it < 4; ++it) {
        int idx = tid + it * 128;
        int r = idx >> 4, c4 = idx & 15;
        __bf16* d = &Bs[buf][r * SB_G + c4 * 4];
        d[0] = (__bf16)brv[it].x; d[1] = (__bf16)brv[it].y;
        d[2] = (__bf16)brv[it].z; d[3] = (__bf16)brv[it].w;
      }
    } else {
#pragma unroll
      for (int it = 0; it < 16; ++it) {
        int idx = tid + it * 128;
        int r = idx >> 6, c = idx & 63;
        Bs[buf][r * SB_G + c] = (__bf16)bsc[it];
      }
    }
  };

  load_tile(0);
  for (int kt = 0; kt < K; kt += 32) {
    const int buf = (kt >> 5) & 1;
    store_tile(buf);
    __syncthreads();
    if (kt + 32 < K) load_tile(kt + 32);  // issue next tile's loads, no wait

    v16bf a0 = make_afrag(&As[buf][(wm + ln) * SA_G], hlf);
    v16bf a1 = make_afrag(&As[buf][(wm + 16 + ln) * SA_G], hlf);
    v16bf b0 = make_bfrag(&Bs[buf][wn + ln], SB_G, hlf);
    v16bf b1 = make_bfrag(&Bs[buf][wn + 16 + ln], SB_G, hlf);

    acc[0][0] = __builtin_amdgcn_wmma_f32_16x16x32_bf16(false, a0, false, b0, (short)0, acc[0][0], false, false);
    acc[0][1] = __builtin_amdgcn_wmma_f32_16x16x32_bf16(false, a0, false, b1, (short)0, acc[0][1], false, false);
    acc[1][0] = __builtin_amdgcn_wmma_f32_16x16x32_bf16(false, a1, false, b0, (short)0, acc[1][0], false, false);
    acc[1][1] = __builtin_amdgcn_wmma_f32_16x16x32_bf16(false, a1, false, b1, (short)0, acc[1][1], false, false);
  }

#pragma unroll
  for (int i = 0; i < 2; ++i) {
#pragma unroll
    for (int j = 0; j < 2; ++j) {
      int n = n0 + wn + j * 16 + ln;
      float bvv = bias ? bias[n] : 0.0f;
#pragma unroll
      for (int r = 0; r < 8; ++r) {
        int m = m0 + wm + i * 16 + 8 * hlf + r;
        float v = acc[i][j][r] + bvv;
        if (act == 1) v = gelu_tanh(v);
        if (res) v += res[(size_t)m * ldc + n];
        C[(size_t)m * ldc + n] = v;
      }
    }
  }
}

// ---------------------------------------------------------------------------
// LayerNorm: one wave per 256-element row.
// ---------------------------------------------------------------------------
__global__ __launch_bounds__(256)
void layernorm(const float* __restrict__ X, const float* __restrict__ g,
               const float* __restrict__ bta, float* __restrict__ Y) {
  int row = blockIdx.x * 8 + (threadIdx.x >> 5);
  int lane = threadIdx.x & 31;
  const float4* xr = (const float4*)(X + (size_t)row * D_);
  float4 x0 = xr[lane], x1 = xr[lane + 32];
  float s = x0.x + x0.y + x0.z + x0.w + x1.x + x1.y + x1.z + x1.w;
  float mean = wave_sum(s) * (1.0f / D_);
  float dx[8] = {x0.x - mean, x0.y - mean, x0.z - mean, x0.w - mean,
                 x1.x - mean, x1.y - mean, x1.z - mean, x1.w - mean};
  float v = 0.0f;
#pragma unroll
  for (int t = 0; t < 8; ++t) v += dx[t] * dx[t];
  v = wave_sum(v) * (1.0f / D_);
  float rs = rsqrtf(v + EPS_);
  float4 g0 = ((const float4*)g)[lane], g1v = ((const float4*)g)[lane + 32];
  float4 b0 = ((const float4*)bta)[lane], b1v = ((const float4*)bta)[lane + 32];
  float4 y0, y1;
  y0.x = dx[0] * rs * g0.x + b0.x;  y0.y = dx[1] * rs * g0.y + b0.y;
  y0.z = dx[2] * rs * g0.z + b0.z;  y0.w = dx[3] * rs * g0.w + b0.w;
  y1.x = dx[4] * rs * g1v.x + b1v.x; y1.y = dx[5] * rs * g1v.y + b1v.y;
  y1.z = dx[6] * rs * g1v.z + b1v.z; y1.w = dx[7] * rs * g1v.w + b1v.w;
  float4* yr = (float4*)(Y + (size_t)row * D_);
  yr[lane] = y0; yr[lane + 32] = y1;
}

// ---------------------------------------------------------------------------
// qp = q_rel + rel_bias ; crel = br . qp   (one wave per (l, b*T+i) row)
// ---------------------------------------------------------------------------
__global__ __launch_bounds__(256)
void prep_qp(const float* __restrict__ Q, const float* __restrict__ rel_bias,
             const float* __restrict__ br, float* __restrict__ qp,
             float* __restrict__ crel) {
  int row = blockIdx.x * 8 + (threadIdx.x >> 5);
  int lane = threadIdx.x & 31;
  int l = row >> 11;            // / (B*T)
  int t = row & 2047;
  int e = lane * 2;
  float q0 = Q[(size_t)t * D_ + (LA + l) * DK_ + e] + rel_bias[l * DK_ + e];
  float q1 = Q[(size_t)t * D_ + (LA + l) * DK_ + e + 1] + rel_bias[l * DK_ + e + 1];
  qp[((size_t)l * 2048 + t) * DK_ + e] = q0;
  qp[((size_t)l * 2048 + t) * DK_ + e + 1] = q1;
  float p = q0 * br[l * DK_ + e] + q1 * br[l * DK_ + e + 1];
  p = wave_sum(p);
  if (lane == 0) crel[(size_t)l * 2048 + t] = p;
}

// ---------------------------------------------------------------------------
// Scores: per (b,h) block. S = q_eff * k_eff^T via WMMA (Pq/Pk folded for
// absolute heads). Staging uses float4 loads with all loads issued before
// dependent converts/stores. 8 waves; wave w owns a 16x128 strip.
// ---------------------------------------------------------------------------
__global__ __launch_bounds__(256)
void attn_scores(const float* __restrict__ Q, const float* __restrict__ Kc,
                 const float* __restrict__ Pq, const float* __restrict__ Pk,
                 float* __restrict__ S) {
  __shared__ __bf16 qs[T_ * 72];     // [i][e]
  __shared__ __bf16 kts[DK_ * 136];  // [e][j]  (transposed at stage time)
  const int b = blockIdx.x >> 2, h = blockIdx.x & 3;
  const int tid = threadIdx.x;

  // 2048 float4s (q) + (k); 8 iterations, 2-4 independent loads per iteration
  for (int idx = tid; idx < (T_ * DK_) / 4; idx += 256) {
    int i = idx >> 4;
    int e4 = (idx & 15) * 4;
    size_t go = ((size_t)b * T_ + i) * D_ + h * DK_ + e4;
    float4 qv = *(const float4*)(Q + go);
    float4 kv = *(const float4*)(Kc + go);
    if (h < LA) {
      size_t po = ((size_t)h * B_ * T_ + b * T_ + i) * DK_ + e4;
      float4 pq = *(const float4*)(Pq + po);
      float4 pk = *(const float4*)(Pk + po);
      qv.x += pq.x; qv.y += pq.y; qv.z += pq.z; qv.w += pq.w;
      kv.x += pk.x; kv.y += pk.y; kv.z += pk.z; kv.w += pk.w;
    }
    __bf16* qd = &qs[i * 72 + e4];
    qd[0] = (__bf16)qv.x; qd[1] = (__bf16)qv.y;
    qd[2] = (__bf16)qv.z; qd[3] = (__bf16)qv.w;
    kts[(e4 + 0) * 136 + i] = (__bf16)kv.x;
    kts[(e4 + 1) * 136 + i] = (__bf16)kv.y;
    kts[(e4 + 2) * 136 + i] = (__bf16)kv.z;
    kts[(e4 + 3) * 136 + i] = (__bf16)kv.w;
  }
  __syncthreads();

  const int wid = tid >> 5, lane = tid & 31;
  const int hlf = lane >> 4, ln = lane & 15;
  v8f acc[8] = {};
#pragma unroll
  for (int kt = 0; kt < DK_; kt += 32) {
    v16bf a = make_afrag(&qs[(wid * 16 + ln) * 72 + kt], hlf);
#pragma unroll
    for (int jt = 0; jt < 8; ++jt) {
      v16bf bb = make_bfrag(&kts[kt * 136 + jt * 16 + ln], 136, hlf);
      acc[jt] = __builtin_amdgcn_wmma_f32_16x16x32_bf16(false, a, false, bb, (short)0, acc[jt], false, false);
    }
  }
  size_t sbase = (((size_t)b * H_ + h) * T_) * (size_t)T_;
#pragma unroll
  for (int jt = 0; jt < 8; ++jt) {
#pragma unroll
    for (int r = 0; r < 8; ++r) {
      int i = wid * 16 + 8 * hlf + r;
      S[sbase + (size_t)i * T_ + jt * 16 + ln] = acc[jt][r];
    }
  }
}

// ---------------------------------------------------------------------------
// Relative-position streaming pass (the refactored 536MB read-once pass):
// S[b, La+l, i, j] += rel_kernel[l,b,i,j,:] . u[l,b*T+i,:] + crel[l,b*T+i]
// One block per (l,b,i); u row cached in LDS. Rows processed in groups of 4
// so 8 float4 loads are in flight before any reduction begins.
// ---------------------------------------------------------------------------
__global__ __launch_bounds__(256)
void rel_stream(const float* __restrict__ rel, const float* __restrict__ u,
                const float* __restrict__ crel, float* __restrict__ S) {
  __shared__ float us[D_];
  const int bid = blockIdx.x;
  const int l = bid / (B_ * T_);
  const int rem = bid % (B_ * T_);
  const int b = rem >> 7, i = rem & 127;
  const int t = b * T_ + i;
  const int tid = threadIdx.x;

  us[tid] = u[((size_t)l * B_ * T_ + t) * D_ + tid];
  __syncthreads();

  const int wid = tid >> 5, lane = tid & 31;
  const float cr = crel[(size_t)l * B_ * T_ + t];
  const size_t rbase = ((((size_t)l * B_ + b) * T_ + i) * (size_t)T_) * D_;
  const size_t sbase = (((size_t)b * H_ + (LA + l)) * T_ + i) * (size_t)T_;
  float4 u0 = ((const float4*)us)[lane * 2];
  float4 u1 = ((const float4*)us)[lane * 2 + 1];

#pragma unroll
  for (int grp = 0; grp < 4; ++grp) {
    int j0 = wid * 16 + grp * 4;
    float4 ra[8];
#pragma unroll
    for (int q = 0; q < 4; ++q) {
      const float4* rp = (const float4*)(rel + rbase + (size_t)(j0 + q) * D_);
      ra[2 * q]     = rp[lane * 2];
      ra[2 * q + 1] = rp[lane * 2 + 1];
    }
#pragma unroll
    for (int q = 0; q < 4; ++q) {
      float p = ra[2*q].x * u0.x + ra[2*q].y * u0.y +
                ra[2*q].z * u0.z + ra[2*q].w * u0.w +
                ra[2*q+1].x * u1.x + ra[2*q+1].y * u1.y +
                ra[2*q+1].z * u1.z + ra[2*q+1].w * u1.w;
      p = wave_sum(p);
      if (lane == 0) S[sbase + j0 + q] += p + cr;
    }
  }
}

// ---------------------------------------------------------------------------
// Masked scaled softmax, in place. One wave per 128-element row.
// ---------------------------------------------------------------------------
__global__ __launch_bounds__(256)
void softmax_mask(float* __restrict__ S, const unsigned char* __restrict__ mask) {
  int row = blockIdx.x * 8 + (threadIdx.x >> 5);
  int lane = threadIdx.x & 31;
  int b = row >> 9;             // / (H*T)
  int i = row & 127;
  float4 s = ((const float4*)(S + (size_t)row * T_))[lane];
  const unsigned char* m = mask + ((size_t)b * T_ + i) * T_ + lane * 4;
  s.x = m[0] ? s.x * SCALE_ : -1e9f;
  s.y = m[1] ? s.y * SCALE_ : -1e9f;
  s.z = m[2] ? s.z * SCALE_ : -1e9f;
  s.w = m[3] ? s.w * SCALE_ : -1e9f;
  float mx = wave_max(fmaxf(fmaxf(s.x, s.y), fmaxf(s.z, s.w)));
  s.x = __expf(s.x - mx); s.y = __expf(s.y - mx);
  s.z = __expf(s.z - mx); s.w = __expf(s.w - mx);
  float inv = 1.0f / wave_sum(s.x + s.y + s.z + s.w);
  s.x *= inv; s.y *= inv; s.z *= inv; s.w *= inv;
  ((float4*)(S + (size_t)row * T_))[lane] = s;
}

// ---------------------------------------------------------------------------
// o = p @ v per (b,h): 128x64 = (128x128)x(128x64), WMMA, K in 4 steps.
// Output written directly in (b, i, h*DK+e) layout for the Wo GEMM.
// ---------------------------------------------------------------------------
__global__ __launch_bounds__(256)
void attn_out(const float* __restrict__ P, const float* __restrict__ V,
              float* __restrict__ O) {
  __shared__ __bf16 ps[T_ * 136];  // [i][j]
  __shared__ __bf16 vs[T_ * 72];   // [j][e]
  const int b = blockIdx.x >> 2, h = blockIdx.x & 3;
  const int tid = threadIdx.x;
  const size_t pbase = (((size_t)b * H_ + h) * T_) * (size_t)T_;

  for (int idx = tid; idx < (T_ * T_) / 4; idx += 256) {
    int i = idx >> 5;
    int j4 = (idx & 31) * 4;
    float4 pv = *(const float4*)(P + pbase + (size_t)i * T_ + j4);
    __bf16* d = &ps[i * 136 + j4];
    d[0] = (__bf16)pv.x; d[1] = (__bf16)pv.y;
    d[2] = (__bf16)pv.z; d[3] = (__bf16)pv.w;
  }
  for (int idx = tid; idx < (T_ * DK_) / 4; idx += 256) {
    int j = idx >> 4;
    int e4 = (idx & 15) * 4;
    float4 vv = *(const float4*)(V + ((size_t)b * T_ + j) * D_ + h * DK_ + e4);
    __bf16* d = &vs[j * 72 + e4];
    d[0] = (__bf16)vv.x; d[1] = (__bf16)vv.y;
    d[2] = (__bf16)vv.z; d[3] = (__bf16)vv.w;
  }
  __syncthreads();

  const int wid = tid >> 5, lane = tid & 31;
  const int hlf = lane >> 4, ln = lane & 15;
  v8f acc[4] = {};
#pragma unroll
  for (int kt = 0; kt < T_; kt += 32) {
    v16bf a = make_afrag(&ps[(wid * 16 + ln) * 136 + kt], hlf);
#pragma unroll
    for (int et = 0; et < 4; ++et) {
      v16bf bb = make_bfrag(&vs[kt * 72 + et * 16 + ln], 72, hlf);
      acc[et] = __builtin_amdgcn_wmma_f32_16x16x32_bf16(false, a, false, bb, (short)0, acc[et], false, false);
    }
  }
#pragma unroll
  for (int et = 0; et < 4; ++et) {
#pragma unroll
    for (int r = 0; r < 8; ++r) {
      int i = wid * 16 + 8 * hlf + r;
      O[((size_t)b * T_ + i) * D_ + h * DK_ + et * 16 + ln] = acc[et][r];
    }
  }
}

// ---------------------------------------------------------------------------
extern "C" void kernel_launch(void* const* d_in, const int* in_sizes, int n_in,
                              void* d_out, int out_size, void* d_ws, size_t ws_size,
                              hipStream_t stream) {
  const float* x          = (const float*)d_in[0];
  const unsigned char* mask = (const unsigned char*)d_in[1];
  const float* abs_kernel = (const float*)d_in[2];
  const float* rel_kernel = (const float*)d_in[3];
  const float* Wq = (const float*)d_in[4];   const float* bq  = (const float*)d_in[5];
  const float* Wk = (const float*)d_in[6];   const float* bk  = (const float*)d_in[7];
  const float* Wv = (const float*)d_in[8];   const float* bv  = (const float*)d_in[9];
  const float* Wpq = (const float*)d_in[10]; const float* bpq = (const float*)d_in[11];
  const float* Wpk = (const float*)d_in[12]; const float* bpk = (const float*)d_in[13];
  const float* Wr  = (const float*)d_in[14]; const float* br  = (const float*)d_in[15];
  const float* rel_bias = (const float*)d_in[16];
  const float* Wo = (const float*)d_in[17];  const float* bo  = (const float*)d_in[18];
  const float* W1 = (const float*)d_in[19];  const float* b1  = (const float*)d_in[20];
  const float* W2 = (const float*)d_in[21];  const float* b2  = (const float*)d_in[22];
  const float* g1 = (const float*)d_in[23];  const float* bl1 = (const float*)d_in[24];
  const float* g2 = (const float*)d_in[25];  const float* bl2 = (const float*)d_in[26];
  float* out = (float*)d_out;
  (void)in_sizes; (void)n_in; (void)out_size; (void)ws_size;

  const int BT = B_ * T_;  // 2048
  float* ws = (float*)d_ws;
  size_t off = 0;
  auto carve = [&](size_t n) { float* p = ws + off; off += n; return p; };
  float* xn   = carve((size_t)BT * D_);
  float* Qb   = carve((size_t)BT * D_);
  float* Kb   = carve((size_t)BT * D_);
  float* Vb   = carve((size_t)BT * D_);
  float* Pq   = carve((size_t)LA * BT * DK_);
  float* Pk   = carve((size_t)LA * BT * DK_);
  float* qp   = carve((size_t)LR * BT * DK_);
  float* crel = carve((size_t)LR * BT);
  float* u    = carve((size_t)LR * BT * D_);
  float* S    = carve((size_t)B_ * H_ * T_ * T_);
  float* O    = carve((size_t)BT * D_);
  float* Xat  = carve((size_t)BT * D_);
  float* xn2  = carve((size_t)BT * D_);
  float* F1   = carve((size_t)BT * DFF_);

  // 1) LN1
  layernorm<<<BT / 8, 256, 0, stream>>>(x, g1, bl1, xn);
  // 2) QKV projections
  gemm_bf16_wmma<true><<<dim3(D_ / 64, BT / 64), 128, 0, stream>>>(xn, D_, Wq, D_, 1, bq, nullptr, Qb, D_, D_, 0);
  gemm_bf16_wmma<true><<<dim3(D_ / 64, BT / 64), 128, 0, stream>>>(xn, D_, Wk, D_, 1, bk, nullptr, Kb, D_, D_, 0);
  gemm_bf16_wmma<true><<<dim3(D_ / 64, BT / 64), 128, 0, stream>>>(xn, D_, Wv, D_, 1, bv, nullptr, Vb, D_, D_, 0);
  // 3) Pq/Pk per absolute layer
  for (int l = 0; l < LA; ++l) {
    gemm_bf16_wmma<true><<<dim3(DK_ / 64, BT / 64), 128, 0, stream>>>(
        abs_kernel + (size_t)l * BT * D_, D_, Wpq + (size_t)l * D_ * DK_, DK_, 1,
        bpq + l * DK_, nullptr, Pq + (size_t)l * BT * DK_, DK_, D_, 0);
    gemm_bf16_wmma<true><<<dim3(DK_ / 64, BT / 64), 128, 0, stream>>>(
        abs_kernel + (size_t)l * BT * D_, D_, Wpk + (size_t)l * D_ * DK_, DK_, 1,
        bpk + l * DK_, nullptr, Pk + (size_t)l * BT * DK_, DK_, D_, 0);
  }
  // 4) qp = q_rel + rel_bias ; crel = br.qp
  prep_qp<<<(LR * BT) / 8, 256, 0, stream>>>(Qb, rel_bias, br, qp, crel);
  // 5) u = qp x Wr^T  (B strided: bks=1, bns=DK)
  for (int l = 0; l < LR; ++l) {
    gemm_bf16_wmma<false><<<dim3(D_ / 64, BT / 64), 128, 0, stream>>>(
        qp + (size_t)l * BT * DK_, DK_, Wr + (size_t)l * D_ * DK_, 1, DK_,
        nullptr, nullptr, u + (size_t)l * BT * D_, D_, DK_, 0);
  }
  // 6) q.k scores (Pq/Pk folded for absolute heads)
  attn_scores<<<B_ * H_, 256, 0, stream>>>(Qb, Kb, Pq, Pk, S);
  // 7) streaming relative term over rel_kernel (memory-bound, read-once)
  rel_stream<<<LR * B_ * T_, 256, 0, stream>>>(rel_kernel, u, crel, S);
  // 8) masked scaled softmax
  softmax_mask<<<(B_ * H_ * T_) / 8, 256, 0, stream>>>(S, mask);
  // 9) o = p @ v
  attn_out<<<B_ * H_, 256, 0, stream>>>(S, Vb, O);
  // 10) Wo projection + residual
  gemm_bf16_wmma<true><<<dim3(D_ / 64, BT / 64), 128, 0, stream>>>(O, D_, Wo, D_, 1, bo, x, Xat, D_, D_, 0);
  // 11) LN2
  layernorm<<<BT / 8, 256, 0, stream>>>(Xat, g2, bl2, xn2);
  // 12) FFN up + GELU
  gemm_bf16_wmma<true><<<dim3(DFF_ / 64, BT / 64), 128, 0, stream>>>(xn2, D_, W1, DFF_, 1, b1, nullptr, F1, DFF_, D_, 1);
  // 13) FFN down + residual -> output
  gemm_bf16_wmma<true><<<dim3(D_ / 64, BT / 64), 128, 0, stream>>>(F1, DFF_, W2, D_, 1, b2, Xat, out, D_, DFF_, 0);
}
